// RNN_88802743812941
// MI455X (gfx1250) — matrix operand order repaired and checked by reading, not compiled, not run on hardware
//
#include <hip/hip_runtime.h>

// ---------------------------------------------------------------------------
// RNN (tanh), H=64, B=512, T=1024.
// Latency-bound sequential scan -> 32 independent waves (16 batch rows each),
// 1 wave per workgroup, no barriers. Recurrent GEMM + output projection done
// with v_wmma_f32_16x16x32_f16 (f32 accumulate), W_hh pinned in VGPRs as f16
// B-operands, h relayout (D-layout -> A-layout) through a padded LDS tile.
// ---------------------------------------------------------------------------

typedef __attribute__((ext_vector_type(16))) _Float16 v16h;
typedef __attribute__((ext_vector_type(8)))  float    v8f;

#define HH   64
#define BB   512
#define TT   1024
#define TM   16          // batch rows per wave (WMMA M)
#define PITCH 68         // LDS row pitch in floats (64+4: conflict-free b64 loads)

#ifndef __has_builtin
#define __has_builtin(x) 0
#endif

__device__ __forceinline__ float fast_tanh(float v) {
#if __has_builtin(__builtin_amdgcn_tanhf)
    return __builtin_amdgcn_tanhf(v);            // v_tanh_f32 (gfx1250 trans op)
#else
    float xc = fminf(fmaxf(v, -15.0f), 15.0f);
    float e  = __builtin_amdgcn_exp2f(xc * 2.8853900817779268f); // 2*log2(e)
    return (e - 1.0f) * __builtin_amdgcn_rcpf(e + 1.0f);
#endif
}

// Load h (16 x 64, f32, row pitch PITCH) from LDS into the two f16 A-operands
// (K = 0..31 and 32..63). A layout (16-bit A 16x32): lane 0-15 -> M=lane,
// VGPR v: K = 16*(v/4) + 8*(lane/16) + 2*(v%4), +1 in high half.
__device__ __forceinline__ void load_A(const float* hsh, int lm, int hf,
                                       v16h& a0, v16h& a1) {
    v16h t0, t1;
#pragma unroll
    for (int v = 0; v < 8; ++v) {
        int k = ((v >> 2) << 4) + (hf << 3) + ((v & 3) << 1);
        const float* row = hsh + lm * PITCH;
        t0[2 * v]     = (_Float16)row[k];
        t0[2 * v + 1] = (_Float16)row[k + 1];
        t1[2 * v]     = (_Float16)row[k + 32];
        t1[2 * v + 1] = (_Float16)row[k + 33];
    }
    a0 = t0; a1 = t1;
}

__global__ __launch_bounds__(32, 1) void rnn_scan_kernel(
    const float* __restrict__ x,      // [B,T]
    const float* __restrict__ h0,     // [B,H]
    const float* __restrict__ W_ih,   // [H]
    const float* __restrict__ b_ih,   // [H]
    const float* __restrict__ W_hh,   // [H,H]
    const float* __restrict__ b_hh,   // [H]
    const float* __restrict__ W_out,  // [H]
    const float* __restrict__ b_out,  // [1]
    float* __restrict__ out)          // out[B*T] ++ hT[B*H]
{
    const int lane = threadIdx.x;         // 0..31 (wave32)
    const int lm   = lane & 15;
    const int hf   = lane >> 4;           // lane half
    const int m0   = blockIdx.x * TM;     // batch tile base

    __shared__ float h_sh[TM * PITCH];

    // ---- init: h0 tile into LDS (f32) -----------------------------------
    for (int i = lane; i < TM * HH; i += 32)
        h_sh[(i / HH) * PITCH + (i % HH)] = h0[m0 * HH + i];

    // ---- loop-invariant B operands: W_hh^T as f16, pinned in VGPRs ------
    // f16 B (32x16) layout: lane L holds N=L%16; element s holds K=8*(L/16)+s
    // (s<8), K=16+8*(L/16)+(s-8) (s>=8).  B[k][n] = W_hh[n_global][k_global].
    v16h bw[4][2];
#pragma unroll
    for (int jb = 0; jb < 4; ++jb)
#pragma unroll
        for (int kc = 0; kc < 2; ++kc) {
            int kg = kc * 32 + 8 * hf;
            v16h tb;
#pragma unroll
            for (int v = 0; v < 8; ++v) {
                tb[v]     = (_Float16)W_hh[(jb * 16 + lm) * HH + kg + v];
                tb[v + 8] = (_Float16)W_hh[(jb * 16 + lm) * HH + kg + 16 + v];
            }
            bw[jb][kc] = tb;
        }

    // W_out as a B operand with only column n==0 non-zero.
    v16h bo[2];
#pragma unroll
    for (int kc = 0; kc < 2; ++kc) {
        v16h tb = {};
        if (lm == 0)
            for (int v = 0; v < 8; ++v) {
                tb[v]     = (_Float16)W_out[kc * 32 + 8 * hf + v];
                tb[v + 8] = (_Float16)W_out[kc * 32 + 16 + 8 * hf + v];
            }
        bo[kc] = tb;
    }

    // Per-lane, per-tile constants: n = jb*16 + lm
    float cbias[4], wih[4];
#pragma unroll
    for (int jb = 0; jb < 4; ++jb) {
        int n = jb * 16 + lm;
        cbias[jb] = b_ih[n] + b_hh[n];
        wih[jb]   = W_ih[n];
    }
    const float bout = b_out[0];

    const float* xrow = x + (m0 + lm) * TT;  // lane L streams batch row L&15

    // ---- sequential scan -------------------------------------------------
    for (int t = 0; t < TT; ++t) {
        // x for this lane's row; broadcast the 8 rows this half needs.
        float xv = xrow[t];
        if (t + 8 < TT) __builtin_prefetch(&xrow[t + 8], 0, 0);
        float xs[8];
#pragma unroll
        for (int r = 0; r < 8; ++r)
            xs[r] = __shfl(xv, r + 8 * hf, 32);

        // A operands = h_{t-1} (f32 LDS -> f16)
        v16h a0, a1;
        load_A(h_sh, lm, hf, a0, a1);

        // Deferred output projection for step t-1 (h_{t-1} == current A).
        if (t > 0) {
            v8f pacc = {};
            pacc = __builtin_amdgcn_wmma_f32_16x16x32_f16(
                false, a0, false, bo[0], (short)0, pacc, false, false);
            pacc = __builtin_amdgcn_wmma_f32_16x16x32_f16(
                false, a1, false, bo[1], (short)0, pacc, false, false);
            if (lm == 0) {
#pragma unroll
                for (int r = 0; r < 8; ++r)
                    out[(m0 + r + 8 * hf) * TT + (t - 1)] = pacc[r] + bout;
            }
        }

        // Accumulators pre-loaded with xp = x*W_ih + (b_ih+b_hh): bias is free.
        v8f acc[4];
#pragma unroll
        for (int jb = 0; jb < 4; ++jb)
#pragma unroll
            for (int r = 0; r < 8; ++r)
                acc[jb][r] = fmaf(xs[r], wih[jb], cbias[jb]);

        // h_{t-1} @ W_hh^T : 4 N-tiles x 2 K-chunks of v_wmma_f32_16x16x32_f16
#pragma unroll
        for (int jb = 0; jb < 4; ++jb) {
            acc[jb] = __builtin_amdgcn_wmma_f32_16x16x32_f16(
                false, a0, false, bw[jb][0], (short)0, acc[jb], false, false);
            acc[jb] = __builtin_amdgcn_wmma_f32_16x16x32_f16(
                false, a1, false, bw[jb][1], (short)0, acc[jb], false, false);
        }

        // tanh, write h_t back to LDS (intra-wave DS is in-order: no barrier).
        // D layout: VGPR r -> M = r + 8*half, lane -> N = jb*16 + lm.
#pragma unroll
        for (int jb = 0; jb < 4; ++jb)
#pragma unroll
            for (int r = 0; r < 8; ++r) {
                float hv = fast_tanh(acc[jb][r]);
                h_sh[(r + 8 * hf) * PITCH + jb * 16 + lm] = hv;
            }
    }

    // ---- epilogue: projection for t = T-1 --------------------------------
    {
        v16h a0, a1;
        load_A(h_sh, lm, hf, a0, a1);
        v8f pacc = {};
        pacc = __builtin_amdgcn_wmma_f32_16x16x32_f16(
            false, a0, false, bo[0], (short)0, pacc, false, false);
        pacc = __builtin_amdgcn_wmma_f32_16x16x32_f16(
            false, a1, false, bo[1], (short)0, pacc, false, false);
        if (lm == 0) {
#pragma unroll
            for (int r = 0; r < 8; ++r)
                out[(m0 + r + 8 * hf) * TT + (TT - 1)] = pacc[r] + bout;
        }
    }

    // ---- hT output: [1,B,H] after out[B*T] -------------------------------
    float* hT = out + BB * TT + m0 * HH;
    for (int i = lane; i < TM * HH; i += 32)
        hT[i] = h_sh[(i / HH) * PITCH + (i % HH)];
}

extern "C" void kernel_launch(void* const* d_in, const int* in_sizes, int n_in,
                              void* d_out, int out_size, void* d_ws, size_t ws_size,
                              hipStream_t stream) {
    (void)in_sizes; (void)n_in; (void)out_size; (void)d_ws; (void)ws_size;
    const float* x     = (const float*)d_in[0];
    const float* h0    = (const float*)d_in[1];
    const float* W_ih  = (const float*)d_in[2];
    const float* b_ih  = (const float*)d_in[3];
    const float* W_hh  = (const float*)d_in[4];
    const float* b_hh  = (const float*)d_in[5];
    const float* W_out = (const float*)d_in[6];
    const float* b_out = (const float*)d_in[7];
    float* out = (float*)d_out;

    dim3 grid(BB / TM);   // 32 independent chains
    dim3 block(32);       // one wave32 per workgroup, no barriers
    rnn_scan_kernel<<<grid, block, 0, stream>>>(
        x, h0, W_ih, b_ih, W_hh, b_hh, W_out, b_out, out);
}
